// BindingFormer_39127152066623
// MI455X (gfx1250) — matrix operand order repaired and checked by reading, not compiled, not run on hardware
//
#include <hip/hip_runtime.h>
#include <hip/hip_bf16.h>

// ---------------- constants ----------------
#define NN        1024
#define DIM       256
#define PDIM      128
#define E_AB_CNT  131072
#define E_TOT     133120          // E_AB_CNT + 2048
#define FEAT      704             // 128 geom + 64 rope + 256 + 256
#define K1TILES   22              // 704 / 32
#define N1TILES   16              // 256 / 16
#define K2TILES   8               // 256 / 32
#define N2TILES   8               // 128 / 16
#define W1F_ELEMS (K1TILES * N1TILES * 32 * 16)   // 180224
#define W2F_ELEMS (K2TILES * N2TILES * 32 * 16)   // 32768
#define MTILES    4               // 64 edges per block = 4 x 16-row M tiles
#define EPB       (MTILES * 16)   // 64 edges per block

typedef __attribute__((ext_vector_type(16))) __bf16 v16bf;
typedef __attribute__((ext_vector_type(8)))  float  v8f;
typedef __attribute__((ext_vector_type(4)))  unsigned int v4u;

union FragBF { v16bf v; v4u q[2]; };

__device__ __forceinline__ unsigned short f2bf(float x) {
    unsigned int u = __float_as_uint(x);
    unsigned int r = u + 0x7FFFu + ((u >> 16) & 1u);
    return (unsigned short)(r >> 16);
}

// branch-free tanh: native V_TANH_F32 on gfx1250 if the builtin exists,
// else hardware-exp based identity (still branch-free).
__device__ __forceinline__ float fast_tanh(float x) {
#if __has_builtin(__builtin_amdgcn_tanhf)
    return __builtin_amdgcn_tanhf(x);
#else
    float e = __expf(2.0f * x);
    return (e - 1.0f) / (e + 1.0f);
#endif
}

__device__ __forceinline__ float gelu_f(float x) {
    float x3 = x * x * x;
    return 0.5f * x * (1.0f + fast_tanh(0.7978845608028654f * (x + 0.044715f * x3)));
}

// ---------------- weight prep: f32 -> bf16 fragment-native layout ----------------
// B-fragment layout (16-bit, 32x16): lane l, element j (j=0..15):
//   k_local = 8*(l>>4) + (j&7) + 16*(j>>3) ; n_local = l & 15
__global__ __launch_bounds__(256) void prep_kernel(const float* __restrict__ w1e,
                                                   const float* __restrict__ w2e,
                                                   unsigned short* __restrict__ w1f,
                                                   unsigned short* __restrict__ w2f) {
    int idx = blockIdx.x * 256 + threadIdx.x;
    if (idx < W1F_ELEMS) {
        int j  = idx & 15;
        int l  = (idx >> 4) & 31;
        int f  = idx >> 9;
        int kb = f >> 4, nb = f & 15;
        int k  = kb * 32 + 8 * (l >> 4) + (j & 7) + 16 * (j >> 3);
        int n  = nb * 16 + (l & 15);
        w1f[idx] = f2bf(w1e[k * 256 + n]);
    } else if (idx < W1F_ELEMS + W2F_ELEMS) {
        int i2 = idx - W1F_ELEMS;
        int j  = i2 & 15;
        int l  = (i2 >> 4) & 31;
        int f  = i2 >> 9;
        int kb = f >> 3, nb = f & 7;
        int k  = kb * 32 + 8 * (l >> 4) + (j & 7) + 16 * (j >> 3);
        int n  = nb * 16 + (l & 15);
        w2f[i2] = f2bf(w2e[k * 128 + n]);
    }
}

// ---------------- node kernel: frames + residue MLP (rep -> bf16) ----------------
__global__ __launch_bounds__(256) void node_kernel(const float* __restrict__ emb,
                                                   const float* __restrict__ tstep,
                                                   const float* __restrict__ coords,
                                                   const float* __restrict__ tfreq,
                                                   const float* __restrict__ w1r,
                                                   const float* __restrict__ b1r,
                                                   const float* __restrict__ w2r,
                                                   const float* __restrict__ b2r,
                                                   unsigned short* __restrict__ rep,
                                                   float* __restrict__ RT) {
    __shared__ float sx[288];
    __shared__ float sh[256];
    const int nid = blockIdx.x;
    const int t   = threadIdx.x;

    sx[t] = emb[(size_t)nid * 256 + t];
    if (t < 32) {
        float tv  = tstep[nid];
        int   f   = t & 15;
        float ang = 6.283185307179586f * tv * tfreq[f];
        sx[256 + t] = (t < 16) ? __sinf(ang) : __cosf(ang);
    }
    if (t == 0) {
        const float* cc = coords + (size_t)nid * 9;
        float ax = cc[0], ay = cc[1], az = cc[2];
        float bx = cc[3], by = cc[4], bz = cc[5];
        float cx = cc[6], cy = cc[7], cz = cc[8];
        float v1x = cx - bx, v1y = cy - by, v1z = cz - bz;
        float v2x = ax - bx, v2y = ay - by, v2z = az - bz;
        float n1  = sqrtf(v1x * v1x + v1y * v1y + v1z * v1z) + 1e-6f;
        float e1x = v1x / n1, e1y = v1y / n1, e1z = v1z / n1;
        float dp  = e1x * v2x + e1y * v2y + e1z * v2z;
        float u2x = v2x - e1x * dp, u2y = v2y - e1y * dp, u2z = v2z - e1z * dp;
        float n2  = sqrtf(u2x * u2x + u2y * u2y + u2z * u2z) + 1e-6f;
        float e2x = u2x / n2, e2y = u2y / n2, e2z = u2z / n2;
        float e3x = e1y * e2z - e1z * e2y;
        float e3y = e1z * e2x - e1x * e2z;
        float e3z = e1x * e2y - e1y * e2x;
        float* rt = RT + (size_t)nid * 12;
        rt[0] = e1x; rt[1] = e2x; rt[2] = e3x;
        rt[3] = e1y; rt[4] = e2y; rt[5] = e3y;
        rt[6] = e1z; rt[7] = e2z; rt[8] = e3z;
        rt[9] = bx;  rt[10] = by; rt[11] = bz;
    }
    __syncthreads();

    float acc = b1r[t];
    #pragma unroll 8
    for (int k = 0; k < 288; ++k) acc += sx[k] * w1r[k * 256 + t];
    sh[t] = gelu_f(acc);
    __syncthreads();

    float acc2 = b2r[t];
    #pragma unroll 8
    for (int k = 0; k < 256; ++k) acc2 += sh[k] * w2r[k * 256 + t];
    acc2 += sx[t];                       // residual = res_embedding_in
    rep[(size_t)nid * 256 + t] = f2bf(acc2);
}

// ---------------- edge kernel: featurize + 2-layer WMMA MLP ----------------
// 64 edges per block (4 M-tiles), 4 waves; each wave owns 4 (layer1) / 2 (layer2)
// N-tiles and reuses every B fragment across all 4 M-tiles.
__global__ __launch_bounds__(128) void edge_kernel(const int* __restrict__ ga,
                                                   const int* __restrict__ gb,
                                                   const int* __restrict__ pid,
                                                   const int* __restrict__ ridx,
                                                   const int* __restrict__ chain,
                                                   const float* __restrict__ cfreq,
                                                   const float* __restrict__ RT,
                                                   const float* __restrict__ wgeom,
                                                   const unsigned short* __restrict__ rep,
                                                   const unsigned short* __restrict__ w1f,
                                                   const float* __restrict__ b1e,
                                                   const unsigned short* __restrict__ w2f,
                                                   const float* __restrict__ b2e,
                                                   float* __restrict__ out) {
    __shared__ __align__(16) unsigned short s_in[EPB][FEAT];   // 90112 B
    __shared__ __align__(16) unsigned short s_hid[EPB][256];   // 32768 B
    __shared__ float s_geomv[EPB][27];                          // 6912 B
    __shared__ int   s_src[EPB], s_dst[EPB];
    __shared__ float s_disp[EPB], s_rsc[EPB];

    const int t = threadIdx.x;

    // ---- phase 1a: per-edge scalars + 27-d geometry vector (threads 0..63) ----
    if (t < EPB) {
        int ge = blockIdx.x * EPB + t;
        int s, d;
        if (ge < E_AB_CNT) {
            s = ga[ge]; d = gb[ge];
        } else {
            int i  = ge - E_AB_CNT;
            int bb = i >> 10;
            int rm = i & 1023;
            s = pid[bb * 32 + (rm >> 5)];
            d = pid[bb * 32 + (rm & 31)];
        }
        s_src[t] = s; s_dst[t] = d;
        float disp = ((float)ridx[s] - (float)ridx[d]) * 0.125f;
        float mask = (chain[s] == chain[d]) ? 1.0f : 0.0f;
        s_disp[t] = disp;
        s_rsc[t]  = mask / (fabsf(disp) + 1.0f);

        const float* Rs = RT + (size_t)s * 12;
        const float* Rd = RT + (size_t)d * 12;
        float dx = Rd[9] - Rs[9], dy = Rd[10] - Rs[10], dz = Rd[11] - Rs[11];
        float dist = sqrtf(dx * dx + dy * dy + dz * dz + 1e-6f);
        const float sig = 20.0f / 15.0f;
        #pragma unroll
        for (int j = 0; j < 15; ++j) {
            float cj = (float)j * (20.0f / 14.0f);
            float z  = (dist - cj) / sig;
            s_geomv[t][j] = __expf(-0.5f * z * z);
        }
        #pragma unroll
        for (int a = 0; a < 3; ++a)
            #pragma unroll
            for (int b = 0; b < 3; ++b)
                s_geomv[t][15 + a * 3 + b] =
                    Rs[a] * Rd[b] + Rs[3 + a] * Rd[3 + b] + Rs[6 + a] * Rd[6 + b];
        float inv = 1.0f / (dist + 1.0f);
        #pragma unroll
        for (int k = 0; k < 3; ++k)
            s_geomv[t][24 + k] = (Rs[k] * dx + Rs[3 + k] * dy + Rs[6 + k] * dz) * inv;
    }
    __syncthreads();

    // ---- phase 1b: build edge_in tile (64 x 704 bf16) ----
    {   // geom projection: col t (0..127) across 64 edges
        float wg[27];
        #pragma unroll
        for (int k = 0; k < 27; ++k) wg[k] = wgeom[k * 128 + t];
        #pragma unroll 4
        for (int e = 0; e < EPB; ++e) {
            float acc = 0.0f;
            #pragma unroll
            for (int k = 0; k < 27; ++k) acc += s_geomv[e][k] * wg[k];
            s_in[e][t] = f2bf(acc);
        }
    }
    {   // rope: 64 values per edge, 2 threads per edge
        int e  = t >> 1;
        int j0 = (t & 1) * 32;
        float disp = s_disp[e], rs = s_rsc[e];
        #pragma unroll
        for (int q = 0; q < 32; ++q) {
            int   f   = j0 + q;
            float ang = 6.283185307179586f * disp * cfreq[f & 31];
            float v   = (f < 32) ? __sinf(ang) : __cosf(ang);
            s_in[e][128 + f] = f2bf(v * rs);
        }
    }
    {   // gather rep[src], rep[dst]: 4096 x 16B chunks
        #pragma unroll
        for (int it = 0; it < 32; ++it) {
            int cid = t + 128 * it;
            int e   = cid >> 6;
            int q   = cid & 63;
            int node = (q < 32) ? s_src[e] : s_dst[e];
            int qq   = q & 31;
            const v4u* sp = (const v4u*)(rep + (size_t)node * 256) + qq;
            v4u* dp = (v4u*)&s_in[e][192 + ((q < 32) ? 0 : 256) + qq * 8];
            *dp = *sp;
        }
    }
    __syncthreads();

    const int lane = t & 31;
    const int wv   = t >> 5;
    const int half = lane >> 4;
    const int mrow = lane & 15;

    // ---- layer 1: edge_in(64x704) @ w1(704x256) ----
    v8f acc1[MTILES][4] = {};
    #pragma unroll 1
    for (int kb = 0; kb < K1TILES; ++kb) {
        FragBF a[MTILES];
        #pragma unroll
        for (int mt = 0; mt < MTILES; ++mt) {
            const v4u* ap = (const v4u*)&s_in[mt * 16 + mrow][kb * 32 + 8 * half];
            a[mt].q[0] = ap[0];
            a[mt].q[1] = ap[2];            // +16 elements
        }
        #pragma unroll
        for (int nt = 0; nt < 4; ++nt) {
            int nb = wv * 4 + nt;
            FragBF b;
            const v4u* bp = (const v4u*)(w1f + ((size_t)(kb * 16 + nb) * 32 + lane) * 16);
            b.q[0] = bp[0];
            b.q[1] = bp[1];
            #pragma unroll
            for (int mt = 0; mt < MTILES; ++mt)
                acc1[mt][nt] = __builtin_amdgcn_wmma_f32_16x16x32_bf16(
                    false, a[mt].v, false, b.v, (short)0, acc1[mt][nt], false, false);
        }
    }
    #pragma unroll
    for (int nt = 0; nt < 4; ++nt) {
        int nb   = wv * 4 + nt;
        int ncol = nb * 16 + mrow;
        float bias = b1e[ncol];
        #pragma unroll
        for (int mt = 0; mt < MTILES; ++mt)
            #pragma unroll
            for (int r = 0; r < 8; ++r) {
                int m = mt * 16 + r + 8 * half;
                s_hid[m][ncol] = f2bf(gelu_f(acc1[mt][nt][r] + bias));
            }
    }
    __syncthreads();

    // ---- layer 2: hidden(64x256) @ w2(256x128) ----
    v8f acc2[MTILES][2] = {};
    #pragma unroll 1
    for (int kb = 0; kb < K2TILES; ++kb) {
        FragBF a[MTILES];
        #pragma unroll
        for (int mt = 0; mt < MTILES; ++mt) {
            const v4u* ap = (const v4u*)&s_hid[mt * 16 + mrow][kb * 32 + 8 * half];
            a[mt].q[0] = ap[0];
            a[mt].q[1] = ap[2];
        }
        #pragma unroll
        for (int nt = 0; nt < 2; ++nt) {
            int nb = wv * 2 + nt;
            FragBF b;
            const v4u* bp = (const v4u*)(w2f + ((size_t)(kb * 8 + nb) * 32 + lane) * 16);
            b.q[0] = bp[0];
            b.q[1] = bp[1];
            #pragma unroll
            for (int mt = 0; mt < MTILES; ++mt)
                acc2[mt][nt] = __builtin_amdgcn_wmma_f32_16x16x32_bf16(
                    false, a[mt].v, false, b.v, (short)0, acc2[mt][nt], false, false);
        }
    }
    size_t base = (size_t)blockIdx.x * EPB;
    #pragma unroll
    for (int nt = 0; nt < 2; ++nt) {
        int nb   = wv * 2 + nt;
        int ncol = nb * 16 + mrow;
        float bias = b2e[ncol];
        #pragma unroll
        for (int mt = 0; mt < MTILES; ++mt)
            #pragma unroll
            for (int r = 0; r < 8; ++r) {
                int m = mt * 16 + r + 8 * half;
                out[(base + m) * 128 + ncol] = acc2[mt][nt][r] + bias;
            }
    }
}

// ---------------- launcher ----------------
extern "C" void kernel_launch(void* const* d_in, const int* in_sizes, int n_in,
                              void* d_out, int out_size, void* d_ws, size_t ws_size,
                              hipStream_t stream) {
    (void)in_sizes; (void)n_in; (void)out_size; (void)ws_size;
    const float* emb    = (const float*)d_in[0];
    const float* tstep  = (const float*)d_in[1];
    const float* coords = (const float*)d_in[2];
    const float* tfreq  = (const float*)d_in[3];
    const float* cfreq  = (const float*)d_in[4];
    const float* w1r    = (const float*)d_in[5];
    const float* b1r    = (const float*)d_in[6];
    const float* w2r    = (const float*)d_in[7];
    const float* b2r    = (const float*)d_in[8];
    const float* wgeom  = (const float*)d_in[9];
    const float* w1e    = (const float*)d_in[10];
    const float* b1e    = (const float*)d_in[11];
    const float* w2e    = (const float*)d_in[12];
    const float* b2e    = (const float*)d_in[13];
    const int*   ridx   = (const int*)d_in[14];
    const int*   chain  = (const int*)d_in[15];
    const int*   ga     = (const int*)d_in[16];
    const int*   gb     = (const int*)d_in[17];
    const int*   pid    = (const int*)d_in[18];
    float*       out    = (float*)d_out;

    char* ws = (char*)d_ws;
    unsigned short* rep = (unsigned short*)(ws);                 // 1024*256*2 = 524288 B
    float*          RT  = (float*)(ws + 524288);                 // 1024*12*4  =  49152 B
    unsigned short* w1f = (unsigned short*)(ws + 573440);        // 180224*2   = 360448 B
    unsigned short* w2f = (unsigned short*)(ws + 933888);        // 32768*2    =  65536 B

    prep_kernel<<<(W1F_ELEMS + W2F_ELEMS) / 256, 256, 0, stream>>>(w1e, w2e, w1f, w2f);
    node_kernel<<<NN, 256, 0, stream>>>(emb, tstep, coords, tfreq, w1r, b1r, w2r, b2r, rep, RT);
    edge_kernel<<<E_TOT / EPB, 128, 0, stream>>>(ga, gb, pid, ridx, chain, cfreq, RT, wgeom,
                                                 rep, w1f, b1e, w2f, b2e, out);
}
